// WikipediaTransformer_45363444580715
// MI455X (gfx1250) — matrix operand order repaired and checked by reading, not compiled, not run on hardware
//
#include <hip/hip_runtime.h>

// ---------------------------------------------------------------------------
// Types
// ---------------------------------------------------------------------------
typedef __bf16 bf16;
typedef bf16  v8bf  __attribute__((ext_vector_type(8)));
typedef bf16  v16bf __attribute__((ext_vector_type(16)));
typedef float v8f   __attribute__((ext_vector_type(8)));
typedef int   v4i   __attribute__((vector_size(16)));   // matches builtin param

#define BM 128
#define BK 32
#define LDSP 40   // padded LDS row (elements): 80B rows keep b128 alignment

enum { F_OUTBF16 = 1, F_GELU = 4, F_CSKIP = 8, F_CKLIM = 16 };

// ---- CDNA5 async global->LDS path (guarded; falls back to reg prefetch) ----
#if defined(__HIP_DEVICE_COMPILE__) && defined(__has_builtin)
#if __has_builtin(__builtin_amdgcn_global_load_async_to_lds_b128)
#define ASYNC_OK 1
#else
#define ASYNC_OK 0
#endif
#else
#define ASYNC_OK 0
#endif

#if ASYNC_OK
__device__ __forceinline__ __attribute__((address_space(3))) v4i*
to_lds(void* p) {
  return (__attribute__((address_space(3))) v4i*)(unsigned)(uintptr_t)p;
}
__device__ __forceinline__ __attribute__((address_space(1))) v4i*
to_glb(const void* p) {
  return (__attribute__((address_space(1))) v4i*)(uintptr_t)p;
}
#endif

template <int N>
__device__ __forceinline__ void wait_async_le() {
#if ASYNC_OK
#if __has_builtin(__builtin_amdgcn_s_wait_asynccnt)
  __builtin_amdgcn_s_wait_asynccnt(N);
#else
  asm volatile("s_wait_asynccnt %0" ::"i"(N));
#endif
#endif
}

// ---------------------------------------------------------------------------
// fp32 -> bf16 conversion (weights)
// ---------------------------------------------------------------------------
__global__ void cvt_f32_bf16(const float* __restrict__ in, bf16* __restrict__ out,
                             long long n) {
  long long i = (long long)blockIdx.x * blockDim.x + threadIdx.x;
  if (i < n) out[i] = (bf16)in[i];
}

// ---------------------------------------------------------------------------
// Embedding + sinusoidal PE:  x = emb[id]*sqrt(D) + pe[pos]
// ---------------------------------------------------------------------------
__global__ void embed_kernel(const int* __restrict__ ids,
                             const float* __restrict__ emb,
                             const float* __restrict__ pe,
                             float* __restrict__ xf, bf16* __restrict__ xb,
                             int total) {
  int t = blockIdx.x * blockDim.x + threadIdx.x;
  if (t >= total) return;
  int d = t % 768;
  int tok = t / 768;
  int pos = tok & 1023;
  float v = emb[(size_t)ids[tok] * 768 + d] * 27.712812921102035f
          + pe[(size_t)pos * 768 + d];
  xf[t] = v;
  xb[t] = (bf16)v;
}

// ---------------------------------------------------------------------------
// Batched bf16 GEMM, v_wmma_f32_16x16x32_bf16.
//   BNK: B stored [N][K] (Q*K^T) instead of [K][N]
//   WNT: wave N-tiles (16 cols each). Block = 8 waves as 4(M)x2(N):
//        tile = 128 x (32*WNT).  Wave tile = 32 x (16*WNT) = 2*WNT WMMAs.
// Double-buffered LDS with software pipeline; A (and B when BNK) staged via
// GLOBAL_LOAD_ASYNC_TO_LDS when available, else register prefetch.
// Batch z -> (z/Hsplit)*sXb + (z%Hsplit)*sXh.
// ---------------------------------------------------------------------------
template <int BNK, int WNT>
__global__ void __launch_bounds__(256)
gemm_bf16_wmma(const bf16* __restrict__ A, long long lda, long long sAb, long long sAh,
               const bf16* __restrict__ Bm, long long ldb, long long sBb, long long sBh,
               void* __restrict__ Cv, long long ldc, long long sCb, long long sCh,
               int Hsplit, int K, const float* __restrict__ bias,
               float alpha, int flags) {
  constexpr int BN_ = 32 * WNT;
  constexpr int ACH = 2;          // A b128 chunks per thread per tile
  constexpr int BCH = BN_ / 64;   // B b128 chunks per thread per tile
  constexpr int NASY = ASYNC_OK ? (ACH + (BNK ? BCH : 0)) : 0;

  const int m0 = blockIdx.y * BM;
  const int n0 = blockIdx.x * BN_;
  if ((flags & F_CSKIP) && n0 > m0 + (BM - 1)) return;

  const int z = blockIdx.z;
  const int zb = z / Hsplit, zh = z % Hsplit;
  A  += zb * sAb + zh * sAh;
  Bm += zb * sBb + zh * sBh;

  int Keff = K;
  if (flags & F_CKLIM) { int kl = m0 + BM; Keff = kl < K ? kl : K; }
  const int nt = Keff / BK;

  __shared__ __align__(16) bf16 lA[2][BM * LDSP];
  __shared__ __align__(16) bf16 lB[2][BN_ * LDSP];

  const int tid = threadIdx.x;
  const int wave = tid >> 5, lane = tid & 31;
  const int wm = wave & 3, wn = wave >> 2;
  const int rsel = lane & 15, khalf = lane >> 4;

  v8f  acc[2][WNT] = {};
  v8bf rA[ACH];          // register prefetch (fallback path / standard B)
  v8bf rB[BCH];

  // ---- stage issue: start global->LDS (async) or global->reg loads --------
  auto issue = [&](int buf, int k0) {
    // A tile: 128x32, k-contiguous chunks
    #pragma unroll
    for (int i = 0; i < ACH; ++i) {
      int c = tid + i * 256;
      int row = c >> 2, off = (c & 3) * 8;
      const bf16* g = A + (size_t)(m0 + row) * lda + k0 + off;
#if ASYNC_OK
      __builtin_amdgcn_global_load_async_to_lds_b128(
          to_glb(g), to_lds(&lA[buf][row * LDSP + off]), 0, 0);
#else
      rA[i] = *(const v8bf*)g;
#endif
    }
    if (BNK) {
      // B stored [N][K]: direct k-contiguous chunks
      #pragma unroll
      for (int i = 0; i < BCH; ++i) {
        int c = tid + i * 256;
        int n = c >> 2, off = (c & 3) * 8;
        const bf16* g = Bm + (size_t)(n0 + n) * ldb + k0 + off;
#if ASYNC_OK
        __builtin_amdgcn_global_load_async_to_lds_b128(
            to_glb(g), to_lds(&lB[buf][n * LDSP + off]), 0, 0);
#else
        rB[i] = *(const v8bf*)g;
#endif
      }
    } else {
      // B stored [K][N]: coalesced rows, transposed at commit
      #pragma unroll
      for (int i = 0; i < BCH; ++i) {
        int c = tid + i * 256;
        int kk = c / (BN_ / 8), n8 = (c % (BN_ / 8)) * 8;
        rB[i] = *(const v8bf*)(Bm + (size_t)(k0 + kk) * ldb + n0 + n8);
      }
    }
  };

  // ---- stage commit: register data -> LDS ---------------------------------
  auto commit = [&](int buf) {
#if !ASYNC_OK
    #pragma unroll
    for (int i = 0; i < ACH; ++i) {
      int c = tid + i * 256;
      int row = c >> 2, off = (c & 3) * 8;
      *(v8bf*)(&lA[buf][row * LDSP + off]) = rA[i];
    }
#endif
    if (BNK) {
#if !ASYNC_OK
      #pragma unroll
      for (int i = 0; i < BCH; ++i) {
        int c = tid + i * 256;
        int n = c >> 2, off = (c & 3) * 8;
        *(v8bf*)(&lB[buf][n * LDSP + off]) = rB[i];
      }
#endif
    } else {
      #pragma unroll
      for (int i = 0; i < BCH; ++i) {
        int c = tid + i * 256;
        int kk = c / (BN_ / 8), n8 = (c % (BN_ / 8)) * 8;
        #pragma unroll
        for (int j = 0; j < 8; ++j) lB[buf][(n8 + j) * LDSP + kk] = rB[i][j];
      }
    }
  };

  issue(0, 0);

  for (int t = 0; t < nt; ++t) {
    const int cur = t & 1;
    commit(cur);                         // (waits reg loads via dependences)
    const bool nxt = (t + 1) < nt;
    if (nxt) issue(cur ^ 1, (t + 1) * BK);
    if (nxt) wait_async_le<NASY>(); else wait_async_le<0>();
    __syncthreads();                     // tile t fully resident in LDS

    // ---- fragments (A 16x32 / B 32x16; both use contiguous k = 16*khalf + s) ----
    v16bf af[2], bfr[WNT];
    #pragma unroll
    for (int mt = 0; mt < 2; ++mt) {
      int row = wm * 32 + mt * 16 + rsel;
      v8bf lo = *(const v8bf*)(&lA[cur][row * LDSP + khalf * 16]);
      v8bf hi = *(const v8bf*)(&lA[cur][row * LDSP + khalf * 16 + 8]);
      v16bf a;
      #pragma unroll
      for (int i = 0; i < 8; ++i) { a[i] = lo[i]; a[8 + i] = hi[i]; }
      af[mt] = a;
    }
    #pragma unroll
    for (int ntile = 0; ntile < WNT; ++ntile) {
      int col = wn * (WNT * 16) + ntile * 16 + rsel;
      bfr[ntile] = *(const v16bf*)(&lB[cur][col * LDSP + khalf * 16]);
    }
    #pragma unroll
    for (int mt = 0; mt < 2; ++mt)
      #pragma unroll
      for (int ntile = 0; ntile < WNT; ++ntile)
        acc[mt][ntile] = __builtin_amdgcn_wmma_f32_16x16x32_bf16(
            false, af[mt], false, bfr[ntile], (short)0, acc[mt][ntile],
            false, false);

    __syncthreads();                     // all waves done reading buf `cur`
  }

  // ---- epilogue -----------------------------------------------------------
  float* Cf = (float*)Cv;
  bf16*  Cb = (bf16*)Cv;
  const size_t cbase = (size_t)zb * sCb + (size_t)zh * sCh;
  const int rhi = lane >> 4;
  #pragma unroll
  for (int mt = 0; mt < 2; ++mt) {
    #pragma unroll
    for (int ntile = 0; ntile < WNT; ++ntile) {
      int col = n0 + wn * (WNT * 16) + ntile * 16 + rsel;
      float bv = bias ? bias[col] : 0.0f;
      #pragma unroll
      for (int r = 0; r < 8; ++r) {
        int row = m0 + wm * 32 + mt * 16 + r + rhi * 8;
        float v = acc[mt][ntile][r] * alpha + bv;
        if (flags & F_GELU) v = 0.5f * v * (1.0f + erff(v * 0.70710678118654752f));
        size_t idx = cbase + (size_t)row * ldc + col;
        if (flags & F_OUTBF16) Cb[idx] = (bf16)v; else Cf[idx] = v;
      }
    }
  }
}

// ---------------------------------------------------------------------------
// Fused (residual add) + LayerNorm: one wave per row of 768.
// ---------------------------------------------------------------------------
__global__ void __launch_bounds__(256)
addln_kernel(const float* __restrict__ resid, const float* __restrict__ add,
             const float* __restrict__ g, const float* __restrict__ bta,
             float* __restrict__ outf, bf16* __restrict__ outb, int rows) {
  int r = (int)((blockIdx.x * blockDim.x + threadIdx.x) >> 5);
  int lane = threadIdx.x & 31;
  if (r >= rows) return;
  const float* r0 = resid + (size_t)r * 768;
  const float* a0 = add ? add + (size_t)r * 768 : nullptr;
  float s = 0.0f, sq = 0.0f;
  for (int d = lane; d < 768; d += 32) {
    float v = r0[d] + (a0 ? a0[d] : 0.0f);
    s += v; sq += v * v;
  }
  #pragma unroll
  for (int m = 16; m >= 1; m >>= 1) {
    s  += __shfl_xor(s,  m, 32);
    sq += __shfl_xor(sq, m, 32);
  }
  float mu  = s * (1.0f / 768.0f);
  float var = sq * (1.0f / 768.0f) - mu * mu;
  float rs  = rsqrtf(var + 1e-5f);
  for (int d = lane; d < 768; d += 32) {
    float v = r0[d] + (a0 ? a0[d] : 0.0f);
    float o = (v - mu) * rs * g[d] + bta[d];
    if (outf) outf[(size_t)r * 768 + d] = o;
    if (outb) outb[(size_t)r * 768 + d] = (bf16)o;
  }
}

// ---------------------------------------------------------------------------
// Causal softmax: one wave per row of S=1024; bf16 probs, zeros above diag.
// ---------------------------------------------------------------------------
__global__ void __launch_bounds__(256)
softmax_causal(float* __restrict__ scores, bf16* __restrict__ probs, int rows) {
  int r = (int)((blockIdx.x * blockDim.x + threadIdx.x) >> 5);
  int lane = threadIdx.x & 31;
  if (r >= rows) return;
  int i = r & 1023;
  int len = i + 1;
  float* sp = scores + (size_t)r * 1024;
  bf16*  pp = probs  + (size_t)r * 1024;
  float mx = -1e30f;
  for (int j = lane; j < len; j += 32) mx = fmaxf(mx, sp[j]);
  #pragma unroll
  for (int m = 16; m >= 1; m >>= 1) mx = fmaxf(mx, __shfl_xor(mx, m, 32));
  float sum = 0.0f;
  for (int j = lane; j < len; j += 32) {
    float e = __expf(sp[j] - mx);
    sp[j] = e;
    sum += e;
  }
  #pragma unroll
  for (int m = 16; m >= 1; m >>= 1) sum += __shfl_xor(sum, m, 32);
  float inv = 1.0f / sum;
  for (int j = lane; j < 1024; j += 32)
    pp[j] = (j < len) ? (bf16)(sp[j] * inv) : (bf16)0.0f;
}

// ---------------------------------------------------------------------------
// Host launch
// ---------------------------------------------------------------------------
extern "C" void kernel_launch(void* const* d_in, const int* in_sizes, int n_in,
                              void* d_out, int out_size, void* d_ws, size_t ws_size,
                              hipStream_t stream) {
  (void)in_sizes; (void)n_in; (void)out_size; (void)ws_size;
  const int Bb = 2, S = 1024, D = 768, Hh = 12, DK = 64, L = 6, DFF = 3072, V = 32000;
  const int T = Bb * S;

  const int*   ids    = (const int*)  d_in[0];
  const float* emb    = (const float*)d_in[1];
  const float* pe     = (const float*)d_in[2];
  const float* wq     = (const float*)d_in[3];
  const float* bq     = (const float*)d_in[4];
  const float* wk     = (const float*)d_in[5];
  const float* bk     = (const float*)d_in[6];
  const float* wv     = (const float*)d_in[7];
  const float* bv     = (const float*)d_in[8];
  const float* wo     = (const float*)d_in[9];
  const float* bo     = (const float*)d_in[10];
  const float* ln1g   = (const float*)d_in[11];
  const float* ln1b   = (const float*)d_in[12];
  const float* w1     = (const float*)d_in[13];
  const float* b1     = (const float*)d_in[14];
  const float* w2     = (const float*)d_in[15];
  const float* b2     = (const float*)d_in[16];
  const float* ln2g   = (const float*)d_in[17];
  const float* ln2b   = (const float*)d_in[18];
  const float* lnfg   = (const float*)d_in[19];
  const float* lnfb   = (const float*)d_in[20];
  const float* head_w = (const float*)d_in[21];
  const float* head_b = (const float*)d_in[22];
  float* out = (float*)d_out;

  char* wsp = (char*)d_ws;
  auto alloc = [&](size_t bytes) -> void* {
    void* r = (void*)wsp;
    wsp += (bytes + 255) & ~(size_t)255;
    return r;
  };
  bf16* wq_bf = (bf16*)alloc((size_t)L * D * D * sizeof(bf16));
  bf16* wk_bf = (bf16*)alloc((size_t)L * D * D * sizeof(bf16));
  bf16* wv_bf = (bf16*)alloc((size_t)L * D * D * sizeof(bf16));
  bf16* wo_bf = (bf16*)alloc((size_t)L * D * D * sizeof(bf16));
  bf16* w1_bf = (bf16*)alloc((size_t)L * D * DFF * sizeof(bf16));
  bf16* w2_bf = (bf16*)alloc((size_t)L * DFF * D * sizeof(bf16));
  bf16* hw_bf = (bf16*)alloc((size_t)D * V * sizeof(bf16));
  float* x_f  = (float*)alloc((size_t)T * D * sizeof(float));
  bf16*  x_b  = (bf16*) alloc((size_t)T * D * sizeof(bf16));
  bf16*  q_b  = (bf16*) alloc((size_t)T * D * sizeof(bf16));
  bf16*  k_b  = (bf16*) alloc((size_t)T * D * sizeof(bf16));
  bf16*  v_b  = (bf16*) alloc((size_t)T * D * sizeof(bf16));
  float* sc   = (float*)alloc((size_t)Bb * Hh * S * S * sizeof(float));
  bf16*  pr   = (bf16*) alloc((size_t)Bb * Hh * S * S * sizeof(bf16));
  bf16*  o_b  = (bf16*) alloc((size_t)T * D * sizeof(bf16));
  float* t0   = (float*)alloc((size_t)T * D * sizeof(float));
  bf16*  f1_b = (bf16*) alloc((size_t)T * DFF * sizeof(bf16));

  auto cvt = [&](const float* src, bf16* dst, long long n) {
    cvt_f32_bf16<<<dim3((unsigned)((n + 255) / 256)), dim3(256), 0, stream>>>(src, dst, n);
  };
  cvt(wq, wq_bf, (long long)L * D * D);
  cvt(wk, wk_bf, (long long)L * D * D);
  cvt(wv, wv_bf, (long long)L * D * D);
  cvt(wo, wo_bf, (long long)L * D * D);
  cvt(w1, w1_bf, (long long)L * D * DFF);
  cvt(w2, w2_bf, (long long)L * DFF * D);
  cvt(head_w, hw_bf, (long long)D * V);

  {
    int total = T * D;
    embed_kernel<<<dim3((total + 255) / 256), dim3(256), 0, stream>>>(
        ids, emb, pe, x_f, x_b, total);
  }

  // launch helpers: WNT=4 (BN=128) for big-N, WNT=2 (BN=64) for attn*V
  auto gemm4 = [&](const bf16* A, long long lda, long long sAb, long long sAh,
                   const bf16* Bm, long long ldb, long long sBb, long long sBh,
                   void* C, long long ldc, long long sCb, long long sCh,
                   int Hs, int batch, int M, int N, int K,
                   const float* bias, float alpha, int flags, bool bnk) {
    dim3 grid((unsigned)(N / 128), (unsigned)(M / BM), (unsigned)batch);
    if (bnk)
      gemm_bf16_wmma<1, 4><<<grid, dim3(256), 0, stream>>>(
          A, lda, sAb, sAh, Bm, ldb, sBb, sBh, C, ldc, sCb, sCh, Hs, K, bias, alpha, flags);
    else
      gemm_bf16_wmma<0, 4><<<grid, dim3(256), 0, stream>>>(
          A, lda, sAb, sAh, Bm, ldb, sBb, sBh, C, ldc, sCb, sCh, Hs, K, bias, alpha, flags);
  };
  auto gemm2 = [&](const bf16* A, long long lda, long long sAb, long long sAh,
                   const bf16* Bm, long long ldb, long long sBb, long long sBh,
                   void* C, long long ldc, long long sCb, long long sCh,
                   int Hs, int batch, int M, int N, int K,
                   const float* bias, float alpha, int flags) {
    dim3 grid((unsigned)(N / 64), (unsigned)(M / BM), (unsigned)batch);
    gemm_bf16_wmma<0, 2><<<grid, dim3(256), 0, stream>>>(
        A, lda, sAb, sAh, Bm, ldb, sBb, sBh, C, ldc, sCb, sCh, Hs, K, bias, alpha, flags);
  };
  auto addln = [&](const float* resid, const float* add, const float* g,
                   const float* b, float* of, bf16* ob) {
    addln_kernel<<<dim3(T * 32 / 256), dim3(256), 0, stream>>>(resid, add, g, b, of, ob, T);
  };

  const long long sQKVb = (long long)S * D;
  const long long sSb   = (long long)Hh * S * S;
  const long long sSh   = (long long)S * S;

  for (int l = 0; l < L; ++l) {
    const bf16* Wq = wq_bf + (size_t)l * D * D;
    const bf16* Wk = wk_bf + (size_t)l * D * D;
    const bf16* Wv = wv_bf + (size_t)l * D * D;
    const bf16* Wo = wo_bf + (size_t)l * D * D;
    const bf16* W1 = w1_bf + (size_t)l * D * DFF;
    const bf16* W2 = w2_bf + (size_t)l * DFF * D;

    gemm4(x_b, D, 0, 0, Wq, D, 0, 0, q_b, D, 0, 0, 1, 1, T, D, D,
          bq + l * D, 1.0f, F_OUTBF16, false);
    gemm4(x_b, D, 0, 0, Wk, D, 0, 0, k_b, D, 0, 0, 1, 1, T, D, D,
          bk + l * D, 1.0f, F_OUTBF16, false);
    gemm4(x_b, D, 0, 0, Wv, D, 0, 0, v_b, D, 0, 0, 1, 1, T, D, D,
          bv + l * D, 1.0f, F_OUTBF16, false);

    // scores = (1/8) Q K^T per (b,h): B is [N=key][K=d] -> BNK; skip masked blocks
    gemm4(q_b, D, sQKVb, DK, k_b, D, sQKVb, DK,
          sc, S, sSb, sSh, Hh, Bb * Hh, S, S, DK,
          nullptr, 0.125f, F_CSKIP, true);

    softmax_causal<<<dim3(Bb * Hh * S * 32 / 256), dim3(256), 0, stream>>>(
        sc, pr, Bb * Hh * S);

    // O = P V per (b,h): N = 64 -> WNT=2; K limited to m0+128 (P zero beyond)
    gemm2(pr, S, sSb, sSh, v_b, D, sQKVb, DK,
          o_b, D, sQKVb, DK, Hh, Bb * Hh, S, DK, S,
          nullptr, 1.0f, F_OUTBF16 | F_CKLIM);

    gemm4(o_b, D, 0, 0, Wo, D, 0, 0, t0, D, 0, 0, 1, 1, T, D, D,
          bo + l * D, 1.0f, 0, false);

    addln(x_f, t0, ln1g + l * D, ln1b + l * D, x_f, x_b);

    gemm4(x_b, D, 0, 0, W1, DFF, 0, 0, f1_b, DFF, 0, 0, 1, 1, T, DFF, D,
          b1 + l * DFF, 1.0f, F_OUTBF16 | F_GELU, false);

    gemm4(f1_b, DFF, 0, 0, W2, D, 0, 0, t0, D, 0, 0, 1, 1, T, D, DFF,
          b2 + l * D, 1.0f, 0, false);

    addln(x_f, t0, ln2g + l * D, ln2b + l * D, x_f, x_b);
  }

  addln(x_f, nullptr, lnfg, lnfb, nullptr, x_b);

  gemm4(x_b, D, 0, 0, hw_bf, V, 0, 0, out, V, 0, 0, 1, 1, T, V, D,
        head_b, 1.0f, 0, false);
}
